// RouteGPT_1726576853393
// MI455X (gfx1250) — matrix operand order repaired and checked by reading, not compiled, not run on hardware
//
#include <hip/hip_runtime.h>

// ---------------------------------------------------------------------------
// GPT forward for MI455X (gfx1250, wave32, WMMA bf16 + Tensor Data Mover).
// Memory-bound on weights (~0.2GB bf16 per pass @23.3TB/s ~ 10us floor; all
// per-GEMM operands fit the 192MB L2 so re-reads are L2 hits). Weights are
// converted to bf16 each launch; every matmul (projections, FFN, logits, and
// both attention GEMMs) runs on v_wmma_f32_16x16x32_bf16 with fp32
// accumulation; flash-style attention avoids the 0.5GB score tensor; GEMM /
// attention K tiles are staged to LDS by TDM tensor_load_to_lds with
// pad_interval/pad_amount providing the bank-conflict row padding, double-
// buffered and synchronized with s_wait_tensorcnt.
// ---------------------------------------------------------------------------

#define LNUM   8
#define DMODEL 1024
#define HEADS  16
#define TSEQ   1024
#define VOCAB  2048
#define BATCH  8
#define DHEAD  64
#define FFDIM  4096
#define MROWS  (BATCH * TSEQ)   // 8192

typedef __attribute__((ext_vector_type(16))) __bf16        v16bf;
typedef __attribute__((ext_vector_type(8)))  float         v8f;
typedef __attribute__((ext_vector_type(8)))  unsigned int  v8u;
typedef __attribute__((ext_vector_type(4)))  unsigned int  u32x4;
typedef __attribute__((ext_vector_type(8)))  int           i32x8;
typedef __attribute__((ext_vector_type(4)))  int           i32x4;
typedef unsigned short bhalf;  // raw bf16 bits

#if __has_builtin(__builtin_amdgcn_tensor_load_to_lds)
#define HAS_TDM 1
#else
#define HAS_TDM 0
#endif
#if __has_include(<hip/amd_detail/amd_gfx1250_TDM.h>)
#define TDM_ARITY6 1   // therock/clang-23 lane: 6-arg builtin
#else
#define TDM_ARITY6 0   // ROCm 7.2 / clang-22: 5-arg builtin
#endif

__device__ __forceinline__ bhalf f2bf(float f) {
  unsigned int u = __builtin_bit_cast(unsigned int, f);
  u += 0x7fffu + ((u >> 16) & 1u);          // round-to-nearest-even
  return (bhalf)(u >> 16);
}

__device__ __forceinline__ v8f wmma_bf16(v16bf a, v16bf b, v8f c) {
  return __builtin_amdgcn_wmma_f32_16x16x32_bf16(false, a, false, b,
                                                 (short)0, c, false, false);
}

#if HAS_TDM
__device__ __forceinline__ int rfl(unsigned x) {
  return __builtin_amdgcn_readfirstlane((int)x);
}
// Issue one TDM 2D tile load: global (row-major, stride0 elems between rows)
// -> LDS (contiguous rows + pad_amount DWORDs every 2^(padE+1) DWORDs).
// D# field packing per CDNA5 ISA.
__device__ __forceinline__ void tdm_load_2d(unsigned lds_addr, const void* gptr,
                                            int tile_d0, int tile_d1,
                                            int stride0, int tdim0, int tdim1,
                                            int padE, int padAmtM1) {
  const unsigned long long ga = (unsigned long long)(uintptr_t)gptr;
  const unsigned w0 = 1u;                                   // count=1, user D#
  const unsigned w1 = lds_addr;                             // lds_addr[31:0]
  const unsigned w2 = (unsigned)ga;                         // global_addr lo
  const unsigned w3 = (unsigned)((ga >> 32) & 0x1FFFFFFu) | (2u << 30); // type=2
  const u32x4 g0 = {(unsigned)rfl(w0), (unsigned)rfl(w1),
                    (unsigned)rfl(w2), (unsigned)rfl(w3)};
  const int q0 = (1 << 16) | (1 << 20) | (padE << 22) | (padAmtM1 << 25);
  const int q1 = (tdim0 & 0xFFFF) << 16;                    // tensor_dim0 lo
  const int q2 = (int)(((unsigned)tdim0 >> 16) | (((unsigned)tdim1 & 0xFFFFu) << 16));
  const int q3 = (int)(((unsigned)tdim1 >> 16) | ((unsigned)tile_d0 << 16));
  const int q4 = tile_d1 & 0xFFFF;                          // tile_dim2=0
  const int q5 = stride0;                                   // dim0 stride lo32
  const i32x8 g1 = {rfl(q0), rfl(q1), rfl(q2), rfl(q3),
                    rfl(q4), rfl(q5), 0, 0};
  const i32x4 z = {0, 0, 0, 0};
#if TDM_ARITY6
  const i32x8 z8 = {0, 0, 0, 0, 0, 0, 0, 0};
  __builtin_amdgcn_tensor_load_to_lds(g0, g1, z, z, z8, 0);
#else
  __builtin_amdgcn_tensor_load_to_lds(g0, g1, z, z, 0);
#endif
}
#endif  // HAS_TDM

// ---------------------------------------------------------------------------
// fp32 -> bf16 bulk convert (weights)
// ---------------------------------------------------------------------------
__global__ __launch_bounds__(256) void cvt_f32_bf16(const float* __restrict__ in,
                                                    bhalf* __restrict__ out, int n) {
  const int stride = gridDim.x * blockDim.x;
  for (int i = blockIdx.x * blockDim.x + threadIdx.x; i < n; i += stride) {
    if (i + stride < n) __builtin_prefetch(in + i + stride, 0, 1);
    out[i] = f2bf(in[i]);
  }
}

// ---------------------------------------------------------------------------
// x[m,d] = tok_emb[idx[m],d] + pos_emb[m%T,d]   (fp32 residual stream)
// ---------------------------------------------------------------------------
__global__ __launch_bounds__(256) void embed_kernel(const int* __restrict__ idx,
                                                    const float* __restrict__ tok,
                                                    const float* __restrict__ pos,
                                                    float* __restrict__ x) {
  const int m = blockIdx.x;
  const int t = m & (TSEQ - 1);
  const size_t tokoff = (size_t)idx[m] * DMODEL;
  for (int d = threadIdx.x; d < DMODEL; d += 256)
    x[(size_t)m * DMODEL + d] = tok[tokoff + d] + pos[(size_t)t * DMODEL + d];
}

// ---------------------------------------------------------------------------
// LayerNorm: fp32 in -> bf16 out (GEMM operand)
// ---------------------------------------------------------------------------
__global__ __launch_bounds__(256) void layernorm_bf16(const float* __restrict__ x,
                                                      const float* __restrict__ g,
                                                      const float* __restrict__ b,
                                                      bhalf* __restrict__ out) {
  __shared__ float s1[256], s2[256];
  const int m = blockIdx.x, tid = threadIdx.x;
  const float* row = x + (size_t)m * DMODEL;
  float s = 0.f, ss = 0.f;
  for (int d = tid; d < DMODEL; d += 256) { float v = row[d]; s += v; ss += v * v; }
  s1[tid] = s; s2[tid] = ss;
  __syncthreads();
  for (int o = 128; o > 0; o >>= 1) {
    if (tid < o) { s1[tid] += s1[tid + o]; s2[tid] += s2[tid + o]; }
    __syncthreads();
  }
  const float mu   = s1[0] * (1.0f / DMODEL);
  const float var  = s2[0] * (1.0f / DMODEL) - mu * mu;
  const float rstd = rsqrtf(var + 1e-5f);
  for (int d = tid; d < DMODEL; d += 256)
    out[(size_t)m * DMODEL + d] = f2bf((row[d] - mu) * rstd * g[d] + b[d]);
}

// ---------------------------------------------------------------------------
// GEMM: C[M,N] = A[M,K](bf16) x W[K,N](bf16) + bias, with epilogue variants.
// Block = 128 threads (4 waves), tile 64x64, k-step 32.
// A tile [m][k] and B tile [k][n] staged to LDS by TDM (double-buffered, the
// next tile's DMA overlaps this tile's WMMAs); padded row strides (A: 40
// shorts, B: 72 shorts) come from the TDM pad feature.
// EPI: 0 = f32 out + bias            (logits)
//      1 = bf16 out + bias           (Q/K/V projections)
//      2 = f32 out += acc + bias     (residual accumulate: Wo, W2)
//      3 = bf16 out = gelu(acc+bias) (W1)
// ---------------------------------------------------------------------------
template <int EPI>
__global__ __launch_bounds__(128) void gemm_bf16(const bhalf* __restrict__ A,
                                                 const bhalf* __restrict__ W,
                                                 const float* __restrict__ bias,
                                                 float* __restrict__ outF,
                                                 bhalf* __restrict__ outH,
                                                 int K, int N) {
  __shared__ __align__(16) bhalf As[2][64][40];  // [m][k], 80B rows (16 + 4 pad DW)
  __shared__ __align__(16) bhalf Bs[2][32][72];  // [k][n], 144B rows (32 + 4 pad DW)
  const int tid  = threadIdx.x;
  const int lane = tid & 31;
  const int wave = tid >> 5;
  const int m    = lane & 15;
  const int half = lane >> 4;
  const int bm   = blockIdx.y * 64;
  const int bn   = blockIdx.x * 64;

  v8f acc[4] = {};
  const int nk = K >> 5;

#if HAS_TDM
  if (wave == 0) {
    tdm_load_2d((unsigned)(uintptr_t)&As[0][0][0], A + (size_t)bm * K,
                /*tile*/32, 64, /*stride0*/K, /*tdims*/K, MROWS, /*pad*/3, 3);
    tdm_load_2d((unsigned)(uintptr_t)&Bs[0][0][0], W + bn,
                /*tile*/64, 32, /*stride0*/N, /*tdims*/N, K, /*pad*/4, 3);
  }
#endif

  for (int i = 0; i < nk; ++i) {
#if HAS_TDM
    const int cb = i & 1;
    if (wave == 0) {
      if (i + 1 < nk) {
        const int k0 = (i + 1) * 32;
        const int nb = (i + 1) & 1;
        tdm_load_2d((unsigned)(uintptr_t)&As[nb][0][0],
                    A + (size_t)bm * K + k0, 32, 64, K, K, MROWS, 3, 3);
        tdm_load_2d((unsigned)(uintptr_t)&Bs[nb][0][0],
                    W + (size_t)k0 * N + bn, 64, 32, N, N, K, 4, 3);
        __builtin_amdgcn_s_wait_tensorcnt(2);  // tile i resident (in-order)
      } else {
        __builtin_amdgcn_s_wait_tensorcnt(0);
      }
    }
    __syncthreads();
#else
    const int cb = 0;
    const int k0 = i * 32;
    for (int c = tid; c < 256; c += 128) {
      const int r = c >> 2, cc = (c & 3) * 8;
      *(uint4*)&As[0][r][cc] = *(const uint4*)(A + (size_t)(bm + r) * K + k0 + cc);
    }
    for (int c = tid; c < 256; c += 128) {
      const int kr = c >> 3, nc = (c & 7) * 8;
      *(uint4*)&Bs[0][kr][nc] = *(const uint4*)(W + (size_t)(k0 + kr) * N + bn + nc);
    }
    __syncthreads();
#endif

    // A fragment (k = half*16 + v*2, pairs contiguous; same k order as B)
    v8u au;
#pragma unroll
    for (int v = 0; v < 8; ++v) {
      const int k = half * 16 + (v << 1);
      au[v] = *(const unsigned int*)&As[cb][wave * 16 + m][k];
    }
    const v16bf a = __builtin_bit_cast(v16bf, au);
#pragma unroll
    for (int j = 0; j < 4; ++j) {
      // B fragment (k = half*16 + v*2, same k order as A) from natural [k][n] tile
      v8u bu;
#pragma unroll
      for (int v = 0; v < 8; ++v) {
        const int kb = half * 16 + v * 2;
        const int n  = j * 16 + m;
        bu[v] = (unsigned)Bs[cb][kb][n] | ((unsigned)Bs[cb][kb + 1][n] << 16);
      }
      acc[j] = wmma_bf16(a, __builtin_bit_cast(v16bf, bu), acc[j]);
    }
    __syncthreads();
  }

  // Epilogue. C-layout: row = v + 8*half, col = n (lane&15) within 16x16 tile.
#pragma unroll
  for (int j = 0; j < 4; ++j) {
#pragma unroll
    for (int v = 0; v < 8; ++v) {
      const int row = bm + wave * 16 + half * 8 + v;
      const int col = bn + j * 16 + m;
      float val = acc[j][v] + bias[col];
      const size_t o = (size_t)row * N + col;
      if constexpr (EPI == 0) {
        outF[o] = val;
      } else if constexpr (EPI == 1) {
        outH[o] = f2bf(val);
      } else if constexpr (EPI == 2) {
        outF[o] += val;
      } else {  // exact GELU: 0.5*x*(1+erf(x/sqrt(2)))
        val = 0.5f * val * (1.0f + erff(val * 0.70710678118654752f));
        outH[o] = f2bf(val);
      }
    }
  }
}

// ---------------------------------------------------------------------------
// Flash attention, causal. grid (B*H, T/64), 128 threads = 4 waves.
// Wave w owns 16 query rows; 32-key tiles staged cooperatively (K via TDM,
// V transposed manually). S = Q K^T via WMMA (DH=64 -> 2 k-steps), online
// softmax with cross-lane shfl reductions, O += P V via WMMA with P
// re-staged through LDS into A-fragment layout.
// ---------------------------------------------------------------------------
__global__ __launch_bounds__(128) void attn_flash(const bhalf* __restrict__ Q,
                                                  const bhalf* __restrict__ Kc,
                                                  const bhalf* __restrict__ Vc,
                                                  bhalf* __restrict__ Y) {
  __shared__ __align__(16) bhalf Ks[32][72];      // [key][dh], TDM-padded rows
  __shared__ __align__(16) bhalf VsT[64][40];     // [dh][key]
  __shared__ __align__(16) bhalf Ps[4][16][40];   // per-wave P staging
  const int bh = blockIdx.x, b = bh >> 4, h = bh & 15;
  const int qb = blockIdx.y * 64;
  const int tid = threadIdx.x, lane = tid & 31, wave = tid >> 5;
  const int m = lane & 15, half = lane >> 4;
  const int qrow0 = qb + wave * 16;

  // Loop-invariant Q fragments (two 32-wide k-steps over DH=64)
  v16bf qf[2];
  {
    const size_t qbase = (size_t)(b * TSEQ + qrow0 + m) * DMODEL + h * DHEAD;
#pragma unroll
    for (int s = 0; s < 2; ++s) {
      v8u u;
#pragma unroll
      for (int v = 0; v < 8; ++v) {
        const int k = s * 32 + half * 16 + (v << 1);
        u[v] = *(const unsigned int*)(Q + qbase + k);
      }
      qf[s] = __builtin_bit_cast(v16bf, u);
    }
  }

  v8f O[4] = {};
  float mrun[8], lrun[8];
#pragma unroll
  for (int v = 0; v < 8; ++v) { mrun[v] = -1e30f; lrun[v] = 0.f; }

  const int ntiles = qb / 32 + 2;  // causal: keys up to qb+63
  for (int kt = 0; kt < ntiles; ++kt) {
    const size_t gk = (size_t)(b * TSEQ + kt * 32) * DMODEL + h * DHEAD;
#if HAS_TDM
    if (wave == 0)
      tdm_load_2d((unsigned)(uintptr_t)&Ks[0][0], Kc + gk,
                  /*tile*/64, 32, /*stride0*/DMODEL, DMODEL, MROWS, /*pad*/4, 3);
    for (int c = tid; c < 256; c += 128) {  // V transposed (TDM can't transpose)
      const int kr = c >> 3, cc = (c & 7) * 8;
      uint4 dv = *(const uint4*)(Vc + gk + (size_t)kr * DMODEL + cc);
      const bhalf* e = (const bhalf*)&dv;
#pragma unroll
      for (int ii = 0; ii < 8; ++ii) VsT[cc + ii][kr] = e[ii];
    }
    if (wave == 0) __builtin_amdgcn_s_wait_tensorcnt(0);
    __syncthreads();
#else
    for (int c = tid; c < 256; c += 128) {
      const int kr = c >> 3, cc = (c & 7) * 8;
      const size_t g = gk + (size_t)kr * DMODEL + cc;
      *(uint4*)&Ks[kr][cc] = *(const uint4*)(Kc + g);
      uint4 dv = *(const uint4*)(Vc + g);
      const bhalf* e = (const bhalf*)&dv;
#pragma unroll
      for (int ii = 0; ii < 8; ++ii) VsT[cc + ii][kr] = e[ii];
    }
    __syncthreads();
#endif

    if (kt * 32 <= qrow0 + 15) {  // wave-uniform: EXEC all-ones inside
      // S tiles: keys [kt*32, kt*32+16) and [kt*32+16, kt*32+32)
      v8f s0 = {}, s1 = {};
#pragma unroll
      for (int s = 0; s < 2; ++s) {
        v8u u0, u1;
#pragma unroll
        for (int v = 0; v < 8; ++v) {
          const int kb = s * 32 + half * 16 + v * 2;  // dh pair (B-layout)
          u0[v] = *(const unsigned int*)&Ks[m][kb];
          u1[v] = *(const unsigned int*)&Ks[16 + m][kb];
        }
        s0 = wmma_bf16(qf[s], __builtin_bit_cast(v16bf, u0), s0);
        s1 = wmma_bf16(qf[s], __builtin_bit_cast(v16bf, u1), s1);
      }

      // Causal mask + online softmax (row r = qrow0 + v + 8*half)
      float p0[8], p1[8], alpha[8];
#pragma unroll
      for (int v = 0; v < 8; ++v) {
        const int r  = qrow0 + half * 8 + v;
        const int c0 = kt * 32 + m, c1 = c0 + 16;
        float a0 = (c0 <= r) ? s0[v] * 0.125f : -1e30f;  // 1/sqrt(64)
        float a1 = (c1 <= r) ? s1[v] * 0.125f : -1e30f;
        float mx = fmaxf(a0, a1);
#pragma unroll
        for (int o = 1; o < 16; o <<= 1) mx = fmaxf(mx, __shfl_xor(mx, o, 32));
        const float mn = fmaxf(mrun[v], mx);
        const float e0 = __expf(a0 - mn), e1 = __expf(a1 - mn);
        float sm = e0 + e1;
#pragma unroll
        for (int o = 1; o < 16; o <<= 1) sm += __shfl_xor(sm, o, 32);
        const float al = __expf(mrun[v] - mn);
        lrun[v] = lrun[v] * al + sm;
        mrun[v] = mn;
        alpha[v] = al; p0[v] = e0; p1[v] = e1;
      }
#pragma unroll
      for (int j = 0; j < 4; ++j)
#pragma unroll
        for (int v = 0; v < 8; ++v) O[j][v] *= alpha[v];

      // Re-stage P (C-layout) into LDS, reread in A-fragment layout
#pragma unroll
      for (int v = 0; v < 8; ++v) {
        const int r = half * 8 + v;
        Ps[wave][r][m]      = f2bf(p0[v]);
        Ps[wave][r][16 + m] = f2bf(p1[v]);
      }
      v8u pu;
#pragma unroll
      for (int v = 0; v < 8; ++v) {
        const int k = half * 16 + (v << 1);
        pu[v] = *(const unsigned int*)&Ps[wave][m][k];
      }
      const v16bf pf = __builtin_bit_cast(v16bf, pu);

      // O += P (16x32) x V (32x64)
#pragma unroll
      for (int j = 0; j < 4; ++j) {
        v8u vu;
#pragma unroll
        for (int v = 0; v < 8; ++v)
          vu[v] = *(const unsigned int*)&VsT[j * 16 + m][half * 16 + v * 2];
        O[j] = wmma_bf16(pf, __builtin_bit_cast(v16bf, vu), O[j]);
      }
    }
    __syncthreads();
  }

  // Y = O / l  (bf16, feeds the Wo GEMM)
#pragma unroll
  for (int j = 0; j < 4; ++j)
#pragma unroll
    for (int v = 0; v < 8; ++v) {
      const int r = qrow0 + half * 8 + v;
      Y[(size_t)(b * TSEQ + r) * DMODEL + h * DHEAD + j * 16 + m] =
          f2bf(O[j][v] / lrun[v]);
    }
}

// ---------------------------------------------------------------------------
// Orchestration
// ---------------------------------------------------------------------------
extern "C" void kernel_launch(void* const* d_in, const int* in_sizes, int n_in,
                              void* d_out, int out_size, void* d_ws, size_t ws_size,
                              hipStream_t stream) {
  (void)in_sizes; (void)n_in; (void)out_size; (void)ws_size;
  const int*   idx   = (const int*)d_in[0];
  const float* tok   = (const float*)d_in[1];
  const float* pos   = (const float*)d_in[2];
  const float* ln1_g = (const float*)d_in[3];
  const float* ln1_b = (const float*)d_in[4];
  const float* Wq    = (const float*)d_in[5];
  const float* bq    = (const float*)d_in[6];
  const float* Wk    = (const float*)d_in[7];
  const float* bk    = (const float*)d_in[8];
  const float* Wv    = (const float*)d_in[9];
  const float* bv    = (const float*)d_in[10];
  const float* Wo    = (const float*)d_in[11];
  const float* bo    = (const float*)d_in[12];
  const float* ln2_g = (const float*)d_in[13];
  const float* ln2_b = (const float*)d_in[14];
  const float* W1    = (const float*)d_in[15];
  const float* b1    = (const float*)d_in[16];
  const float* W2    = (const float*)d_in[17];
  const float* b2    = (const float*)d_in[18];
  const float* lnf_g = (const float*)d_in[19];
  const float* lnf_b = (const float*)d_in[20];
  const float* Wh    = (const float*)d_in[21];
  const float* bh    = (const float*)d_in[22];

  char* ws = (char*)d_ws;
  size_t off = 0;
  auto alloc = [&](size_t bytes) -> char* {
    char* p = ws + off;
    off += (bytes + 255) & ~(size_t)255;
    return p;
  };
  const size_t DD = (size_t)DMODEL * DMODEL;
  bhalf* bWq = (bhalf*)alloc(LNUM * DD * 2);
  bhalf* bWk = (bhalf*)alloc(LNUM * DD * 2);
  bhalf* bWv = (bhalf*)alloc(LNUM * DD * 2);
  bhalf* bWo = (bhalf*)alloc(LNUM * DD * 2);
  bhalf* bW1 = (bhalf*)alloc((size_t)LNUM * DMODEL * FFDIM * 2);
  bhalf* bW2 = (bhalf*)alloc((size_t)LNUM * FFDIM * DMODEL * 2);
  bhalf* bWh = (bhalf*)alloc((size_t)DMODEL * VOCAB * 2);
  float* xf  = (float*)alloc((size_t)MROWS * DMODEL * 4);
  bhalf* xn  = (bhalf*)alloc((size_t)MROWS * DMODEL * 2);
  bhalf* qh  = (bhalf*)alloc((size_t)MROWS * DMODEL * 2);
  bhalf* kh  = (bhalf*)alloc((size_t)MROWS * DMODEL * 2);
  bhalf* vh  = (bhalf*)alloc((size_t)MROWS * DMODEL * 2);
  bhalf* yat = (bhalf*)alloc((size_t)MROWS * DMODEL * 2);
  bhalf* ffh = (bhalf*)alloc((size_t)MROWS * FFDIM * 2);

  // Weight conversion (each launch; kernel_launch must be stateless)
  auto cvt = [&](const float* s, bhalf* d, size_t n) {
    cvt_f32_bf16<<<4096, 256, 0, stream>>>(s, d, (int)n);
  };
  cvt(Wq, bWq, LNUM * DD);
  cvt(Wk, bWk, LNUM * DD);
  cvt(Wv, bWv, LNUM * DD);
  cvt(Wo, bWo, LNUM * DD);
  cvt(W1, bW1, (size_t)LNUM * DMODEL * FFDIM);
  cvt(W2, bW2, (size_t)LNUM * FFDIM * DMODEL);
  cvt(Wh, bWh, (size_t)DMODEL * VOCAB);

  embed_kernel<<<MROWS, 256, 0, stream>>>(idx, tok, pos, xf);

  const dim3 gD(DMODEL / 64, MROWS / 64);   // N=1024 outputs
  const dim3 gF(FFDIM / 64, MROWS / 64);    // N=4096 outputs
  const dim3 gV(VOCAB / 64, MROWS / 64);    // N=2048 outputs
  const dim3 gA(BATCH * HEADS, TSEQ / 64);  // attention

  for (int l = 0; l < LNUM; ++l) {
    layernorm_bf16<<<MROWS, 256, 0, stream>>>(xf, ln1_g + l * DMODEL,
                                              ln1_b + l * DMODEL, xn);
    gemm_bf16<1><<<gD, 128, 0, stream>>>(xn, bWq + l * DD, bq + l * DMODEL,
                                         nullptr, qh, DMODEL, DMODEL);
    gemm_bf16<1><<<gD, 128, 0, stream>>>(xn, bWk + l * DD, bk + l * DMODEL,
                                         nullptr, kh, DMODEL, DMODEL);
    gemm_bf16<1><<<gD, 128, 0, stream>>>(xn, bWv + l * DD, bv + l * DMODEL,
                                         nullptr, vh, DMODEL, DMODEL);
    attn_flash<<<gA, 128, 0, stream>>>(qh, kh, vh, yat);
    gemm_bf16<2><<<gD, 128, 0, stream>>>(yat, bWo + l * DD, bo + l * DMODEL,
                                         xf, nullptr, DMODEL, DMODEL);
    layernorm_bf16<<<MROWS, 256, 0, stream>>>(xf, ln2_g + l * DMODEL,
                                              ln2_b + l * DMODEL, xn);
    gemm_bf16<3><<<gF, 128, 0, stream>>>(xn, bW1 + (size_t)l * DMODEL * FFDIM,
                                         b1 + l * FFDIM, nullptr, ffh,
                                         DMODEL, FFDIM);
    gemm_bf16<2><<<gD, 128, 0, stream>>>(ffh, bW2 + (size_t)l * FFDIM * DMODEL,
                                         b2 + l * DMODEL, xf, nullptr,
                                         FFDIM, DMODEL);
  }

  layernorm_bf16<<<MROWS, 256, 0, stream>>>(xf, lnf_g, lnf_b, xn);
  gemm_bf16<0><<<gV, 128, 0, stream>>>(xn, bWh, bh, (float*)d_out, nullptr,
                                       DMODEL, VOCAB);
}
